// GemmaLavaAttention_80358838108478
// MI455X (gfx1250) — matrix operand order repaired and checked by reading, not compiled, not run on hardware
//
#include <hip/hip_runtime.h>
#include <hip/hip_bf16.h>

// ---------------------------------------------------------------------------
// GemmaLavaAttention on MI455X (gfx1250, wave32, WMMA, async-to-LDS)
// B=2, S=2048, H=2048, NH=32, HD=64
// Pipeline:
//   1) cvt: fp32 -> bf16 for X, Wq, Wk, Wv, Wo (workspace)
//   2) gemm_bf16<1>: Q = X@Wq^T with fused RoPE  -> bf16 [b,h,s,d]
//      gemm_bf16<1>: K = X@Wk^T with fused RoPE  -> bf16 [b,h,s,d]
//      gemm_bf16<2>: V = X@Wv^T                  -> bf16 [b,h,s,d]
//   3) attn_fa: flash attention (online softmax) -> bf16 [b,s,h*64+d]
//   4) gemm_bf16<0>: out = attn@Wo^T             -> fp32 d_out
// GEMM uses double-buffered GLOBAL_LOAD_ASYNC_TO_LDS_B128 staging (ASYNCcnt).
// ---------------------------------------------------------------------------

typedef __bf16 bf16;
typedef __attribute__((ext_vector_type(16))) __bf16 v16bf;
typedef __attribute__((ext_vector_type(8)))  float  v8f;

#define B_   2
#define S_   2048
#define H_   2048
#define NH_  32
#define HD_  64
#define M_   (B_ * S_)

union FragAB { uint4 u4[2]; v16bf v; };
union Pack32 { uint4 u4[4]; __bf16 h[32]; };

__device__ __forceinline__ v8f wmma_bf16(v16bf a, v16bf b, v8f c) {
  // 8 args: (neg_a, A, neg_b, B, c_mod, C, reuse_a, reuse_b)
  return __builtin_amdgcn_wmma_f32_16x16x32_bf16(false, a, false, b,
                                                 (short)0, c, false, false);
}

// Async copy 16B global -> LDS per lane (GV mode, VDST = LDS byte address).
// Flat-aperture rule: generic LDS pointer's low 32 bits are the LDS offset.
__device__ __forceinline__ void async_ld_b128(void* lds, const void* gp) {
  uint32_t l = (uint32_t)(uintptr_t)lds;
  asm volatile("global_load_async_to_lds_b128 %0, %1, off"
               :: "v"(l), "v"(gp) : "memory");
}
__device__ __forceinline__ void wait_async0() {
  asm volatile("s_wait_asynccnt 0x0" ::: "memory");
}

// ---------------------------------------------------------------------------
// fp32 -> bf16 conversion (grid-stride)
// ---------------------------------------------------------------------------
__global__ void cvt_f32_bf16(const float* __restrict__ s,
                             bf16* __restrict__ d, int n) {
  int i = blockIdx.x * blockDim.x + threadIdx.x;
  int stride = gridDim.x * blockDim.x;
  for (; i < n; i += stride) d[i] = (bf16)s[i];
}

// ---------------------------------------------------------------------------
// Tiled GEMM: Y[m,n] = sum_k A[m,k] * W[n,k]   (A: MxK, W: NxK, both bf16)
// Block = 256 threads (8 waves); tile 128x128, K-step 32, double-buffered
// async-to-LDS staging. Wave grid 4x2; per-wave 2x4 fragments of 16x16.
// MODE 0: fp32 row-major out; MODE 1: RoPE -> bf16 [b,h,s,d]; MODE 2: bf16.
// ---------------------------------------------------------------------------
template <int MODE>
__global__ __launch_bounds__(256, 2)
void gemm_bf16(const bf16* __restrict__ A, const bf16* __restrict__ W,
               const float* __restrict__ cosT, const float* __restrict__ sinT,
               bf16* __restrict__ outB, float* __restrict__ outF,
               int Ndim, int Kdim) {
  __shared__ bf16 As[2][128 * 32];
  __shared__ bf16 Ws[2][128 * 32];

  const int tid   = threadIdx.x;
  const int m0    = blockIdx.x * 128;
  const int n0    = blockIdx.y * 128;
  const int w     = tid >> 5;
  const int lane  = tid & 31;
  const int nlane = lane & 15;
  const int half  = lane >> 4;
  const int m_off = (w & 3) * 32;
  const int n_off = (w >> 2) * 64;

  const int ldrow = tid >> 1;          // 0..127
  const int ldc   = (tid & 1) * 16;    // 0 or 16 (each thread: 16 bf16 = 2x b128)

  v8f acc[2][4];
#pragma unroll
  for (int mi = 0; mi < 2; mi++)
#pragma unroll
    for (int ni = 0; ni < 4; ni++)
      acc[mi][ni] = (v8f){0.f, 0.f, 0.f, 0.f, 0.f, 0.f, 0.f, 0.f};

  // Stage tile for k-offset into buffer `buf` (async, no wait here).
  auto stage = [&](int buf, int k0) {
    const bf16* ga = &A[(size_t)(m0 + ldrow) * Kdim + k0 + ldc];
    bf16* la = &As[buf][ldrow * 32 + ldc];
    async_ld_b128(la,     ga);
    async_ld_b128(la + 8, ga + 8);
    const bf16* gw = &W[(size_t)(n0 + ldrow) * Kdim + k0 + ldc];
    bf16* lw = &Ws[buf][ldrow * 32 + ldc];
    async_ld_b128(lw,     gw);
    async_ld_b128(lw + 8, gw + 8);
  };

  stage(0, 0);
  wait_async0();
  __syncthreads();

  int cur = 0;
  for (int k0 = 0; k0 < Kdim; k0 += 32) {
    if (k0 + 32 < Kdim) stage(cur ^ 1, k0 + 32);  // prefetch next tile (async)

    // --- fragment loads per documented gfx1250 WMMA layouts ---
    v16bf a[2], b[4];
#pragma unroll
    for (int mi = 0; mi < 2; mi++) {
      const int row = m_off + mi * 16 + nlane;  // A: M = lane%16
      FragAB f;
      f.u4[0] = *(const uint4*)&As[cur][row * 32 + half * 8];       // K=half*8+0..7
      f.u4[1] = *(const uint4*)&As[cur][row * 32 + 16 + half * 8];  // K=16+half*8+0..7
      a[mi] = f.v;
    }
#pragma unroll
    for (int ni = 0; ni < 4; ni++) {
      const int col = n_off + ni * 16 + nlane;  // B: N = lane%16
      FragAB f;
      f.u4[0] = *(const uint4*)&Ws[cur][col * 32 + half * 8];       // K=half*8+0..7
      f.u4[1] = *(const uint4*)&Ws[cur][col * 32 + 16 + half * 8];  // K=16+half*8+0..7
      b[ni] = f.v;
    }
#pragma unroll
    for (int mi = 0; mi < 2; mi++)
#pragma unroll
      for (int ni = 0; ni < 4; ni++)
        acc[mi][ni] = wmma_bf16(a[mi], b[ni], acc[mi][ni]);

    wait_async0();     // next tile resident in LDS
    __syncthreads();   // all waves done reading `cur`, writes to `cur^1` visible
    cur ^= 1;
  }

  // --- epilogue (C layout: M = r + half*8, N = lane%16) ---
#pragma unroll
  for (int mi = 0; mi < 2; mi++) {
#pragma unroll
    for (int ni = 0; ni < 4; ni++) {
#pragma unroll
      for (int r = 0; r < 8; r++) {
        const int mg = m0 + m_off + mi * 16 + r + half * 8;
        const int ng = n0 + n_off + ni * 16 + nlane;
        const float y = acc[mi][ni][r];
        if (MODE == 0) {
          outF[(size_t)mg * Ndim + ng] = y;
        } else {
          const int bb = mg / S_;
          const int ss = mg % S_;
          const int hh = ng >> 6;
          const int dd = ng & 63;
          float val;
          if (MODE == 1) {
            // rotate-half partner (d +/- 32) lives in fragment ni^2,
            // same lane, same element index r.
            const float yp = acc[mi][ni ^ 2][r];
            const float c  = cosT[ss * HD_ + dd];
            const float sn = sinT[ss * HD_ + dd];
            val = (dd < 32) ? (y * c - yp * sn) : (y * c + yp * sn);
          } else {
            val = y;
          }
          outB[((size_t)(bb * NH_ + hh) * S_ + ss) * HD_ + dd] = (bf16)val;
        }
      }
    }
  }
}

// ---------------------------------------------------------------------------
// Flash attention: grid.x = S/64 q-tiles, grid.y = B*NH. 128 threads (4 waves),
// each wave owns 16 q-rows. Key chunks of 64 staged in LDS (K via async copy);
// online softmax; PV via WMMA with V transposed in LDS.
// ---------------------------------------------------------------------------
__global__ __launch_bounds__(128, 4)
void attn_fa(const bf16* __restrict__ Qg, const bf16* __restrict__ Kg,
             const bf16* __restrict__ Vg, const float* __restrict__ maskp,
             bf16* __restrict__ Og) {
  __shared__ bf16 Kl[64 * 64];       // [key][dim]
  __shared__ bf16 Vt[64 * 64];       // [dim][key] (transposed)
  __shared__ bf16 Pl[4 * 16 * 64];   // per-wave P, [row][key_in_chunk]

  const int tid   = threadIdx.x;
  const int w     = tid >> 5;
  const int lane  = tid & 31;
  const int nlane = lane & 15;
  const int half  = lane >> 4;
  const int bh    = blockIdx.y;
  const int b     = bh >> 5;  // / NH
  const int h     = bh & 31;
  const int qbase = blockIdx.x * 64;
  const int q0    = qbase + w * 16;

  // Q A-fragments (16 rows x 64 head dims = two 16x32 frags), from global
  v16bf aq[2];
  {
    const bf16* qrow = &Qg[((size_t)bh * S_ + q0 + nlane) * HD_];
#pragma unroll
    for (int t = 0; t < 2; t++) {
      FragAB f;
      f.u4[0] = *(const uint4*)&qrow[t * 32 + half * 8];
      f.u4[1] = *(const uint4*)&qrow[t * 32 + 16 + half * 8];
      aq[t] = f.v;
    }
  }

  float mrow[8], lrow[8];
#pragma unroll
  for (int r = 0; r < 8; r++) { mrow[r] = -3.0e38f; lrow[r] = 0.f; }
  v8f o[4];
#pragma unroll
  for (int nf = 0; nf < 4; nf++)
    o[nf] = (v8f){0.f, 0.f, 0.f, 0.f, 0.f, 0.f, 0.f, 0.f};

  const int ldrow = tid >> 1;        // 0..63 (key row)
  const int ldc   = (tid & 1) * 32;  // dim 0 or 32
  const float L2E = 1.4426950408889634f;

  for (int kc = 0; kc < S_; kc += 64) {
    // --- stage K chunk via async-to-LDS; V chunk transposed via registers ---
    {
      const bf16* gk = &Kg[((size_t)bh * S_ + kc + ldrow) * HD_ + ldc];
      bf16* lk = &Kl[ldrow * 64 + ldc];
      async_ld_b128(lk,      gk);
      async_ld_b128(lk +  8, gk +  8);
      async_ld_b128(lk + 16, gk + 16);
      async_ld_b128(lk + 24, gk + 24);
      Pack32 pv;
      const uint4* gv = (const uint4*)&Vg[((size_t)bh * S_ + kc + ldrow) * HD_ + ldc];
      pv.u4[0] = gv[0]; pv.u4[1] = gv[1]; pv.u4[2] = gv[2]; pv.u4[3] = gv[3];
#pragma unroll
      for (int j = 0; j < 32; j++) Vt[(ldc + j) * 64 + ldrow] = pv.h[j];
    }
    wait_async0();
    __syncthreads();

    // --- scores: S = Q K^T  (4 x 16-key subtiles, contraction HD=64 = 2 wmma) ---
    v8f s4[4];
#pragma unroll
    for (int sub = 0; sub < 4; sub++) {
      const bf16* krow = &Kl[(sub * 16 + nlane) * 64];
      FragAB f0, f1;
      f0.u4[0] = *(const uint4*)&krow[half * 8];
      f0.u4[1] = *(const uint4*)&krow[16 + half * 8];
      f1.u4[0] = *(const uint4*)&krow[32 + half * 8];
      f1.u4[1] = *(const uint4*)&krow[32 + 16 + half * 8];
      v8f sc = (v8f){0.f, 0.f, 0.f, 0.f, 0.f, 0.f, 0.f, 0.f};
      sc = wmma_bf16(aq[0], f0.v, sc);
      sc = wmma_bf16(aq[1], f1.v, sc);
      const float mv = maskp[b * S_ + kc + sub * 16 + nlane];
#pragma unroll
      for (int e = 0; e < 8; e++) s4[sub][e] = sc[e] * 0.125f + mv;
    }

    // --- online softmax update (rows r+half*8; cols = 16 lanes of the half) ---
#pragma unroll
    for (int r = 0; r < 8; r++) {
      float v = fmaxf(fmaxf(s4[0][r], s4[1][r]), fmaxf(s4[2][r], s4[3][r]));
#pragma unroll
      for (int off = 1; off < 16; off <<= 1)
        v = fmaxf(v, __shfl_xor(v, off, 32));
      const float mnew = fmaxf(mrow[r], v);
      const float sf = __builtin_exp2f((mrow[r] - mnew) * L2E);
      mrow[r] = mnew;
      lrow[r] *= sf;
#pragma unroll
      for (int nf = 0; nf < 4; nf++) o[nf][r] = o[nf][r] * sf;
      float ls = 0.f;
#pragma unroll
      for (int sub = 0; sub < 4; sub++) {
        const float p = __builtin_exp2f((s4[sub][r] - mnew) * L2E);
        ls += p;
        Pl[(w * 16 + r + half * 8) * 64 + sub * 16 + nlane] = (bf16)p;
      }
#pragma unroll
      for (int off = 1; off < 16; off <<= 1)
        ls += __shfl_xor(ls, off, 32);
      lrow[r] += ls;
    }

    // wave-local LDS RAW: P stored in C-layout, re-read in A-layout
    asm volatile("s_wait_dscnt 0x0" ::: "memory");

    // --- O += P V (contraction over 64 keys = 2 x 32-key wmma steps) ---
#pragma unroll
    for (int t = 0; t < 2; t++) {
      const bf16* prow = &Pl[(w * 16 + nlane) * 64];
      FragAB fa;
      fa.u4[0] = *(const uint4*)&prow[t * 32 + half * 8];
      fa.u4[1] = *(const uint4*)&prow[t * 32 + 16 + half * 8];
#pragma unroll
      for (int nf = 0; nf < 4; nf++) {
        const bf16* vrow = &Vt[(nf * 16 + nlane) * 64];
        FragAB fb;
        fb.u4[0] = *(const uint4*)&vrow[t * 32 + half * 8];
        fb.u4[1] = *(const uint4*)&vrow[t * 32 + 16 + half * 8];
        o[nf] = wmma_bf16(fa.v, fb.v, o[nf]);
      }
    }
    __syncthreads();
  }

  // --- normalize and store to [b, s, h*64+d] bf16 ---
#pragma unroll
  for (int nf = 0; nf < 4; nf++) {
#pragma unroll
    for (int r = 0; r < 8; r++) {
      const int srow = q0 + r + half * 8;
      const float val = o[nf][r] / lrow[r];
      Og[((size_t)b * S_ + srow) * H_ + h * HD_ + nf * 16 + nlane] = (bf16)val;
    }
  }
}

// ---------------------------------------------------------------------------
// Host launcher
// ---------------------------------------------------------------------------
extern "C" void kernel_launch(void* const* d_in, const int* in_sizes, int n_in,
                              void* d_out, int out_size, void* d_ws, size_t ws_size,
                              hipStream_t stream) {
  const float* hs    = (const float*)d_in[0];
  const float* maskp = (const float*)d_in[1];
  const float* cosT  = (const float*)d_in[2];
  const float* sinT  = (const float*)d_in[3];
  const float* Wq    = (const float*)d_in[4];
  const float* Wk    = (const float*)d_in[5];
  const float* Wv    = (const float*)d_in[6];
  const float* Wo    = (const float*)d_in[7];
  float* out = (float*)d_out;

  char* ws = (char*)d_ws;
  const size_t MB = 1ull << 20;
  bf16* Xb  = (bf16*)(ws + 0 * MB);    // 16 MB  (4096x2048)
  bf16* Wqb = (bf16*)(ws + 16 * MB);   //  8 MB
  bf16* Wkb = (bf16*)(ws + 24 * MB);   //  8 MB
  bf16* Wvb = (bf16*)(ws + 32 * MB);   //  8 MB
  bf16* Wob = (bf16*)(ws + 40 * MB);   //  8 MB
  bf16* Qb  = (bf16*)(ws + 48 * MB);   // 16 MB  [b,h,s,d]
  bf16* Kb  = (bf16*)(ws + 64 * MB);   // 16 MB
  bf16* Vb  = (bf16*)(ws + 80 * MB);   // 16 MB
  bf16* Ab  = (bf16*)(ws + 96 * MB);   // 16 MB  [b,s,h*64+d]

  const int nX = M_ * H_;       // 8,388,608
  const int nW = H_ * H_;       // 4,194,304
  cvt_f32_bf16<<<8192, 256, 0, stream>>>(hs, Xb, nX);
  cvt_f32_bf16<<<4096, 256, 0, stream>>>(Wq, Wqb, nW);
  cvt_f32_bf16<<<4096, 256, 0, stream>>>(Wk, Wkb, nW);
  cvt_f32_bf16<<<4096, 256, 0, stream>>>(Wv, Wvb, nW);
  cvt_f32_bf16<<<4096, 256, 0, stream>>>(Wo, Wob, nW);

  dim3 gg(M_ / 128, H_ / 128);  // (32, 16)
  gemm_bf16<1><<<gg, 256, 0, stream>>>(Xb, Wqb, cosT, sinT, Qb, nullptr, H_, H_);
  gemm_bf16<1><<<gg, 256, 0, stream>>>(Xb, Wkb, cosT, sinT, Kb, nullptr, H_, H_);
  gemm_bf16<2><<<gg, 256, 0, stream>>>(Xb, Wvb, nullptr, nullptr, Vb, nullptr, H_, H_);

  dim3 ga(S_ / 64, B_ * NH_);   // (32, 64)
  attn_fa<<<ga, 128, 0, stream>>>(Qb, Kb, Vb, maskp, Ab);

  gemm_bf16<0><<<gg, 256, 0, stream>>>(Ab, Wob, nullptr, nullptr, nullptr, out, H_, H_);
}
